// LlamaAttention_59889023975914
// MI455X (gfx1250) — matrix operand order repaired and checked by reading, not compiled, not run on hardware
//
#include <hip/hip_runtime.h>
#include <hip/hip_bf16.h>
#include <math.h>
#include <stdint.h>

#define S_LEN    2048
#define HID      4096
#define NHEADS   32
#define HDIM     128

typedef __attribute__((ext_vector_type(16))) __bf16 v16bf;
typedef __attribute__((ext_vector_type(8)))  __bf16 v8bf;
typedef __attribute__((ext_vector_type(8)))  float  v8f;

union Frag16 { v16bf v; v8bf h[2]; };

static __device__ inline v8bf cvt8(const float4 a, const float4 b) {
    v8bf o;
    o[0]=(__bf16)a.x; o[1]=(__bf16)a.y; o[2]=(__bf16)a.z; o[3]=(__bf16)a.w;
    o[4]=(__bf16)b.x; o[5]=(__bf16)b.y; o[6]=(__bf16)b.z; o[7]=(__bf16)b.w;
    return o;
}

// CDNA5 async copy global -> LDS (ASYNCcnt tracked), 16B per lane.
static __device__ inline void async_load_b128(uint32_t lds_off, const void* gaddr) {
    asm volatile("global_load_async_to_lds_b128 %0, %1, off"
                 :: "v"(lds_off), "v"((unsigned long long)(uintptr_t)gaddr)
                 : "memory");
}
static __device__ inline void wait_async0() {
    asm volatile("s_wait_asynccnt 0x0" ::: "memory");
}

// ---------------------------------------------------------------------------
// C[M,N] = A[M,K] @ W[N,K]^T + bias[N]
// 128 threads (4 waves), 128x128 tile, BK=32, v_wmma_f32_16x16x32_bf16.
// Software pipeline: global loads for slab k+1 prefetched into registers
// while WMMAs consume slab k from LDS.
// ---------------------------------------------------------------------------
template<typename AT>
__global__ __launch_bounds__(128)
void gemm_bias(const AT* __restrict__ A, const float* __restrict__ W,
               const float* __restrict__ bias, float* __restrict__ C,
               int M, int N, int K)
{
    __shared__ __bf16 sA[128][40];    // [m][k], row stride 80B (16B multiple)
    __shared__ __bf16 sB[32][136];    // [k][n], row stride 272B (16B multiple)

    const int tid  = threadIdx.x;
    const int lane = tid & 31;
    const int wave = tid >> 5;
    const int bm0  = blockIdx.y * 128;
    const int bn0  = blockIdx.x * 128;
    const int wm0  = (wave >> 1) * 64;
    const int wn0  = (wave & 1) * 64;
    const int lrow = lane & 15;
    const int kh   = (lane >> 4) * 8;   // ISA 16-bit A layout: lanes 0-15 hold K {0..7,16..23}

    v8f acc[4][4] = {};

    // register prefetch buffers (unused path eliminated by if constexpr use)
    float4 pa[8];
    v8bf   pab[4];
    float4 pb[8];

    auto load_regs = [&](int k0) {
        if constexpr (sizeof(AT) == 4) {
            const float4* srcA = (const float4*)(A + (size_t)(bm0 + tid) * K + k0);
            #pragma unroll
            for (int c = 0; c < 8; ++c) pa[c] = srcA[c];
        } else {
            const v8bf* srcA = (const v8bf*)(A + (size_t)(bm0 + tid) * K + k0);
            #pragma unroll
            for (int c = 0; c < 4; ++c) pab[c] = srcA[c];
        }
        const float4* srcB = (const float4*)(W + (size_t)(bn0 + tid) * K + k0);
        #pragma unroll
        for (int c = 0; c < 8; ++c) pb[c] = srcB[c];
    };

    load_regs(0);

    for (int k0 = 0; k0 < K; k0 += 32) {
        // ---- commit prefetched slab to LDS (bf16)
        if constexpr (sizeof(AT) == 4) {
            #pragma unroll
            for (int c = 0; c < 4; ++c)
                *(v8bf*)&sA[tid][c * 8] = cvt8(pa[2*c], pa[2*c + 1]);
        } else {
            #pragma unroll
            for (int c = 0; c < 4; ++c)
                *(v8bf*)&sA[tid][c * 8] = pab[c];
        }
        #pragma unroll
        for (int c = 0; c < 8; ++c) {
            float4 f = pb[c];
            sB[c*4 + 0][tid] = (__bf16)f.x;
            sB[c*4 + 1][tid] = (__bf16)f.y;
            sB[c*4 + 2][tid] = (__bf16)f.z;
            sB[c*4 + 3][tid] = (__bf16)f.w;
        }
        __syncthreads();

        // ---- prefetch next slab while computing this one
        if (k0 + 32 < K) load_regs(k0 + 32);

        Frag16 afr[4], bfr[4];
        #pragma unroll
        for (int i = 0; i < 4; ++i) {
            afr[i].h[0] = *(const v8bf*)&sA[wm0 + i*16 + lrow][kh];
            afr[i].h[1] = *(const v8bf*)&sA[wm0 + i*16 + lrow][16 + kh];
        }
        #pragma unroll
        for (int j = 0; j < 4; ++j) {
            #pragma unroll
            for (int s = 0; s < 16; ++s)
                bfr[j].v[s] = sB[(s < 8 ? kh + s : 16 + kh + s - 8)][wn0 + j*16 + lrow];
        }
        #pragma unroll
        for (int i = 0; i < 4; ++i)
            #pragma unroll
            for (int j = 0; j < 4; ++j)
                acc[i][j] = __builtin_amdgcn_wmma_f32_16x16x32_bf16(
                    false, afr[i].v, false, bfr[j].v, (short)0, acc[i][j], false, false);
        __syncthreads();
    }

    // ---- epilogue: C layout lane = N col, VGPR r = rows r / r+8
    #pragma unroll
    for (int i = 0; i < 4; ++i) {
        #pragma unroll
        for (int j = 0; j < 4; ++j) {
            const int col = bn0 + wn0 + j*16 + lrow;
            const float b = bias[col];
            #pragma unroll
            for (int r = 0; r < 8; ++r) {
                const int m = bm0 + wm0 + i*16 + r + 8*(lane >> 4);
                C[(size_t)m * N + col] = acc[i][j][r] + b;
            }
        }
    }
}

// ---------------------------------------------------------------------------
// RoPE (interleaved pairs) on Q,K + f32->bf16 conversion of Q,K,V.
// ---------------------------------------------------------------------------
__global__ void rope_convert(const float* __restrict__ Qf, const float* __restrict__ Kf,
                             const float* __restrict__ Vf,
                             __bf16* __restrict__ Qb, __bf16* __restrict__ Kb,
                             __bf16* __restrict__ Vb, const int* __restrict__ total_seq)
{
    const int idx = blockIdx.x * blockDim.x + threadIdx.x;   // [0, S*HID/2)
    const int s  = idx / (HID / 2);
    const int p  = idx % (HID / 2);
    const int hh = p / (HDIM / 2);
    const int i  = p % (HDIM / 2);
    const size_t base = (size_t)s * HID + hh * HDIM + 2 * i;

    const float pos = (float)(*total_seq - S_LEN + s);
    const float inv_freq = __powf(10000.0f, -(float)(2 * i) / (float)HDIM);
    float sn, cs;
    __sincosf(pos * inv_freq, &sn, &cs);

    const float q1 = Qf[base], q2 = Qf[base + 1];
    Qb[base]     = (__bf16)(q1 * cs - q2 * sn);
    Qb[base + 1] = (__bf16)(q1 * sn + q2 * cs);
    const float k1 = Kf[base], k2 = Kf[base + 1];
    Kb[base]     = (__bf16)(k1 * cs - k2 * sn);
    Kb[base + 1] = (__bf16)(k1 * sn + k2 * cs);
    Vb[base]     = (__bf16)Vf[base];
    Vb[base + 1] = (__bf16)Vf[base + 1];
}

// ---------------------------------------------------------------------------
// Flash attention, bf16 WMMA, f32 online softmax.
// Grid: (S/64, NHEADS); 4 waves/block, each wave owns a 16-row query tile.
// Double-buffered K/V staging: V tile copied with CDNA5 async global->LDS
// (ASYNCcnt), K tile transposed manually; staging of chunk t+1 overlaps the
// 16 WMMAs of chunk t.
// ---------------------------------------------------------------------------
__global__ __launch_bounds__(128)
void flash_attn(const __bf16* __restrict__ Qb, const __bf16* __restrict__ Kb,
                const __bf16* __restrict__ Vb, const float* __restrict__ mask,
                __bf16* __restrict__ Ob)
{
    __shared__ __bf16 sK[2][HDIM][40];   // [buf][d][t]  (transposed K chunk)
    __shared__ __bf16 sV[2][32][136];    // [buf][t][d]
    __shared__ __bf16 sP[4][16][40];     // per-wave P tile [s][t]

    const int tid  = threadIdx.x;
    const int lane = tid & 31;
    const int wave = tid >> 5;
    const int h    = blockIdx.y;
    const int s0   = blockIdx.x * 64 + wave * 16;
    const int lrow = lane & 15;
    const int half = lane >> 4;
    const int kh   = half * 8;

    // Q fragments for this wave's 16 rows (4 chunks of K=32 over HDIM=128)
    Frag16 qf[4];
    {
        const __bf16* qrow = Qb + (size_t)(s0 + lrow) * HID + h * HDIM;
        #pragma unroll
        for (int dc = 0; dc < 4; ++dc) {
            qf[dc].h[0] = *(const v8bf*)(qrow + dc*32 + kh);
            qf[dc].h[1] = *(const v8bf*)(qrow + dc*32 + 16 + kh);
        }
    }

    // staging: 4 threads per key row; V via async-to-LDS, K via manual transpose
    auto stage = [&](int buf, int t0) {
        const int tt = tid >> 2;
        const int q4 = (tid & 3) * 32;
        const __bf16* ksrc = Kb + (size_t)(t0 + tt) * HID + h * HDIM + q4;
        const __bf16* vsrc = Vb + (size_t)(t0 + tt) * HID + h * HDIM + q4;
        const uint32_t vdst = (uint32_t)(uintptr_t)&sV[buf][tt][q4];
        #pragma unroll
        for (int c = 0; c < 4; ++c) {
            async_load_b128(vdst + c * 16, vsrc + c * 8);
            v8bf kv = *(const v8bf*)(ksrc + c * 8);
            #pragma unroll
            for (int e = 0; e < 8; ++e) sK[buf][q4 + c*8 + e][tt] = kv[e];
        }
    };

    v8f oacc[8] = {};
    float Mrow[8], Lrow[8];
    #pragma unroll
    for (int r = 0; r < 8; ++r) { Mrow[r] = -__builtin_inff(); Lrow[r] = 0.0f; }

    const float scale = 0.08838834764831845f;  // 1/sqrt(128)

    stage(0, 0);
    wait_async0();
    __syncthreads();

    for (int t0 = 0; t0 < S_LEN; t0 += 32) {
        const int buf = (t0 >> 5) & 1;
        if (t0 + 32 < S_LEN) stage(buf ^ 1, t0 + 32);   // overlap next staging

        // ---- scores: two 16-wide t tiles, each accumulating K=128 over 4 WMMAs
        v8f  sc[2];
        float tmax[2][8];
        #pragma unroll
        for (int tt = 0; tt < 2; ++tt) {
            v8f s8 = {};
            #pragma unroll
            for (int dc = 0; dc < 4; ++dc) {
                Frag16 bfr;
                #pragma unroll
                for (int s = 0; s < 16; ++s)
                    bfr.v[s] = sK[buf][dc*32 + (s < 8 ? kh + s : 16 + kh + s - 8)][tt*16 + lrow];
                s8 = __builtin_amdgcn_wmma_f32_16x16x32_bf16(
                    false, qf[dc].v, false, bfr.v, (short)0, s8, false, false);
            }
            #pragma unroll
            for (int r = 0; r < 8; ++r) {
                const int srow = s0 + r + 8*half;
                const int col  = t0 + tt*16 + lrow;
                float v = s8[r] * scale + mask[(size_t)srow * S_LEN + col];
                s8[r] = v;
                #pragma unroll
                for (int off = 8; off >= 1; off >>= 1)
                    v = fmaxf(v, __shfl_xor(v, off, 32));   // reduce over 16 t-lanes
                tmax[tt][r] = v;
            }
            sc[tt] = s8;
        }

        // ---- online softmax update
        float alpha[8];
        #pragma unroll
        for (int r = 0; r < 8; ++r) {
            const float newM = fmaxf(Mrow[r], fmaxf(tmax[0][r], tmax[1][r]));
            alpha[r] = __expf(Mrow[r] - newM);
            Mrow[r] = newM;
        }
        #pragma unroll
        for (int j = 0; j < 8; ++j)
            #pragma unroll
            for (int r = 0; r < 8; ++r)
                oacc[j][r] *= alpha[r];

        #pragma unroll
        for (int tt = 0; tt < 2; ++tt) {
            #pragma unroll
            for (int r = 0; r < 8; ++r) {
                float p = __expf(sc[tt][r] - Mrow[r]);
                sP[wave][r + 8*half][tt*16 + lrow] = (__bf16)p;
                float ssum = p;
                #pragma unroll
                for (int off = 8; off >= 1; off >>= 1)
                    ssum += __shfl_xor(ssum, off, 32);
                Lrow[r] = (tt == 0 ? Lrow[r] * alpha[r] : Lrow[r]) + ssum;
            }
        }
        __syncthreads();   // P visible across lanes before A-frag reload

        // ---- P @ V: one A-frag (16x32), 8 d-tiles
        Frag16 pf;
        pf.h[0] = *(const v8bf*)&sP[wave][lrow][kh];
        pf.h[1] = *(const v8bf*)&sP[wave][lrow][16 + kh];
        #pragma unroll
        for (int j = 0; j < 8; ++j) {
            Frag16 bfr;
            #pragma unroll
            for (int s = 0; s < 16; ++s)
                bfr.v[s] = sV[buf][(s < 8 ? kh + s : 16 + kh + s - 8)][j*16 + lrow];
            oacc[j] = __builtin_amdgcn_wmma_f32_16x16x32_bf16(
                false, pf.v, false, bfr.v, (short)0, oacc[j], false, false);
        }

        wait_async0();     // next chunk's async V writes landed in LDS
        __syncthreads();   // all waves done staging next / reading current
    }

    // ---- epilogue: normalize and store bf16 attn output
    #pragma unroll
    for (int j = 0; j < 8; ++j) {
        const int d = j*16 + lrow;
        #pragma unroll
        for (int r = 0; r < 8; ++r) {
            const int s = s0 + r + 8*half;
            Ob[(size_t)s * HID + h * HDIM + d] = (__bf16)(oacc[j][r] / Lrow[r]);
        }
    }
}

// ---------------------------------------------------------------------------
extern "C" void kernel_launch(void* const* d_in, const int* in_sizes, int n_in,
                              void* d_out, int out_size, void* d_ws, size_t ws_size,
                              hipStream_t stream)
{
    const float* X    = (const float*)d_in[0];
    const float* mask = (const float*)d_in[1];
    const float* Wq   = (const float*)d_in[2];
    const float* bq   = (const float*)d_in[3];
    const float* Wk   = (const float*)d_in[4];
    const float* bk   = (const float*)d_in[5];
    const float* Wv   = (const float*)d_in[6];
    const float* bv   = (const float*)d_in[7];
    const float* Wo   = (const float*)d_in[8];
    const float* bo   = (const float*)d_in[9];
    const int*   tsl  = (const int*)d_in[10];

    char* ws = (char*)d_ws;
    const size_t SZ_F32 = (size_t)S_LEN * HID * 4;   // 33.5 MB
    const size_t SZ_BF  = (size_t)S_LEN * HID * 2;   // 16.8 MB
    float*  Qf = (float*)(ws);
    float*  Kf = (float*)(ws + SZ_F32);
    float*  Vf = (float*)(ws + 2 * SZ_F32);
    __bf16* Qb = (__bf16*)(ws + 3 * SZ_F32);
    __bf16* Kb = (__bf16*)(ws + 3 * SZ_F32 + SZ_BF);
    __bf16* Vb = (__bf16*)(ws + 3 * SZ_F32 + 2 * SZ_BF);
    __bf16* Ob = (__bf16*)(ws + 3 * SZ_F32 + 3 * SZ_BF);

    const dim3 gemm_grid(HID / 128, S_LEN / 128);   // (32, 16)
    gemm_bias<float><<<gemm_grid, 128, 0, stream>>>(X, Wq, bq, Qf, S_LEN, HID, HID);
    gemm_bias<float><<<gemm_grid, 128, 0, stream>>>(X, Wk, bk, Kf, S_LEN, HID, HID);
    gemm_bias<float><<<gemm_grid, 128, 0, stream>>>(X, Wv, bv, Vf, S_LEN, HID, HID);

    rope_convert<<<(S_LEN * HID / 2) / 256, 256, 0, stream>>>(Qf, Kf, Vf, Qb, Kb, Vb, tsl);

    flash_attn<<<dim3(S_LEN / 64, NHEADS), 128, 0, stream>>>(Qb, Kb, Vb, mask, Ob);

    gemm_bias<__bf16><<<gemm_grid, 128, 0, stream>>>(Ob, Wo, bo, (float*)d_out, S_LEN, HID, HID);
}
